// SlidingWindowAttention_49100066128361
// MI455X (gfx1250) — matrix operand order repaired and checked by reading, not compiled, not run on hardware
//
#include <hip/hip_runtime.h>

typedef __attribute__((ext_vector_type(16))) _Float16 v16h;
typedef __attribute__((ext_vector_type(8)))  float    v8f;
typedef __attribute__((__vector_size__(16))) int      i32x4_;

#define SEQ  4096
#define NH   16
#define DK   64
#define DIM  1024
#define BLK  256
#define NEG_INF (-3.0e38f)

// ---------------------------------------------------------------------------
// Async global->LDS copy (gfx1250 GLOBAL_LOAD_ASYNC_TO_LDS_B128, ASYNCcnt).
// The builtin takes pointers to 128-bit int vectors (per hipcc diagnostic);
// cast chain handles address-space qualification either way.
// Guarded: falls back to a synchronous uint4 copy if the builtin is absent.
// ---------------------------------------------------------------------------
#if defined(__HIP_DEVICE_COMPILE__) && __has_builtin(__builtin_amdgcn_global_load_async_to_lds_b128)
#define ASYNC_COPY_B128(gsrc, ldst)                                            \
  __builtin_amdgcn_global_load_async_to_lds_b128(                              \
      (__attribute__((address_space(1))) i32x4_*)(i32x4_*)(void*)(gsrc),       \
      (__attribute__((address_space(3))) i32x4_*)(i32x4_*)(void*)(ldst),       \
      0, 0)
#define ASYNC_WAIT() asm volatile("s_wait_asynccnt 0x0" ::: "memory")
#else
#define ASYNC_COPY_B128(gsrc, ldst) (*(uint4*)(ldst) = *(const uint4*)(gsrc))
#define ASYNC_WAIT()
#endif

// ---------------------------------------------------------------------------
// WMMA helpers (gfx1250, wave32).
// A-fragment layout for V_WMMA_F32_16X16X32_F16 (16x32 f16, 8 VGPRs/lane):
//   lane: row = lane&15, kbase = (lane>>4)*8
//   elem j<8  -> K = kbase + j ;  elem j>=8 -> K = kbase + 16 + (j-8)
// The B fragment mirrors this with lane = output column, so loading a
// row-major matrix "A-style" gives exactly the B operand of (X * M^T).
// ---------------------------------------------------------------------------
__device__ __forceinline__ v16h frag_load(const _Float16* base, int ld) {
  const int lane = threadIdx.x & 31;
  const int row  = lane & 15;
  const int kb   = (lane >> 4) << 3;
  const _Float16* p = base + row * ld + kb;
  v16h r;
#pragma unroll
  for (int j = 0; j < 8; ++j) { r[j] = p[j]; r[j + 8] = p[16 + j]; }
  return r;
}

__device__ __forceinline__ v8f wmma_f16(v16h a, v16h b, v8f c) {
  return __builtin_amdgcn_wmma_f32_16x16x32_f16(false, a, false, b,
                                                (short)0, c, false, false);
}

__device__ __forceinline__ v8f v8f_zero() {
  v8f z = {0.f, 0.f, 0.f, 0.f, 0.f, 0.f, 0.f, 0.f};
  return z;
}

// ---------------------------------------------------------------------------
// Kernel 1: projection GEMM.  Out[s][e] = sum_d X[s][d]*W[e][d]  (X * W^T),
// written f16 in head-major layout [b][h][s][dk].  Tile 128(M) x 64(N),
// 256 threads = 8 waves, wave tile 32x32 = 2x2 WMMA.
// ---------------------------------------------------------------------------
__global__ void proj_gemm(const float* __restrict__ X, const float* __restrict__ W,
                          _Float16* __restrict__ Out) {
  __shared__ _Float16 As[128][40];
  __shared__ _Float16 Bs[64][40];
  const int t = threadIdx.x, w = t >> 5, lane = t & 31;
  const int ncol = lane & 15, mhalf = (lane >> 4) << 3;
  const int m0 = blockIdx.x * 128, n0 = blockIdx.y * 64;
  const int wm = (w >> 1) * 32, wn = (w & 1) * 32;

  v8f acc[2][2];
#pragma unroll
  for (int i = 0; i < 2; ++i)
#pragma unroll
    for (int j = 0; j < 2; ++j) acc[i][j] = v8f_zero();

  for (int k0 = 0; k0 < DIM; k0 += 32) {
    // A tile: 128x32 fp32 -> f16   (1024 float4, 4 per thread)
#pragma unroll
    for (int i = 0; i < 4; ++i) {
      int u = t + i * 256;
      int row = u >> 3, c4 = (u & 7) * 4;
      float4 d = *(const float4*)(X + (size_t)(m0 + row) * DIM + k0 + c4);
      As[row][c4 + 0] = (_Float16)d.x;  As[row][c4 + 1] = (_Float16)d.y;
      As[row][c4 + 2] = (_Float16)d.z;  As[row][c4 + 3] = (_Float16)d.w;
    }
    // B tile: 64 rows of W (output features) x 32 cols
#pragma unroll
    for (int i = 0; i < 2; ++i) {
      int u = t + i * 256;
      int row = u >> 3, c4 = (u & 7) * 4;
      float4 d = *(const float4*)(W + (size_t)(n0 + row) * DIM + k0 + c4);
      Bs[row][c4 + 0] = (_Float16)d.x;  Bs[row][c4 + 1] = (_Float16)d.y;
      Bs[row][c4 + 2] = (_Float16)d.z;  Bs[row][c4 + 3] = (_Float16)d.w;
    }
    __syncthreads();

    v16h a0 = frag_load(&As[wm][0], 40);
    v16h a1 = frag_load(&As[wm + 16][0], 40);
    v16h b0 = frag_load(&Bs[wn][0], 40);
    v16h b1 = frag_load(&Bs[wn + 16][0], 40);
    acc[0][0] = wmma_f16(a0, b0, acc[0][0]);
    acc[0][1] = wmma_f16(a0, b1, acc[0][1]);
    acc[1][0] = wmma_f16(a1, b0, acc[1][0]);
    acc[1][1] = wmma_f16(a1, b1, acc[1][1]);
    __syncthreads();
  }

  const int h = blockIdx.y;   // 64-wide N tile == one head
#pragma unroll
  for (int i = 0; i < 2; ++i)
#pragma unroll
    for (int j = 0; j < 2; ++j)
#pragma unroll
      for (int r = 0; r < 8; ++r) {
        int m  = m0 + wm + i * 16 + mhalf + r;
        int dk = wn + j * 16 + ncol;
        int bb = m >> 12, s = m & (SEQ - 1);
        Out[(((size_t)(bb * NH + h) * SEQ + s) * DK) + dk] = (_Float16)acc[i][j][r];
      }
}

// ---------------------------------------------------------------------------
// Kernel 2: RoPE in-place on head-major f16 buffer.  Column pair (e, e+512)
// lives at heads (h, h+8) with equal dk;  inv_freq = 10000^(-e/512).
// ---------------------------------------------------------------------------
__global__ void rope_inplace(_Float16* __restrict__ arr) {
  int idx = blockIdx.x * blockDim.x + threadIdx.x;
  if (idx >= 2 * 8 * SEQ * DK) return;
  int dk = idx & 63;
  int s  = (idx >> 6) & (SEQ - 1);
  int h  = (idx >> 18) & 7;
  int b  = idx >> 21;
  int j  = h * DK + dk;
  float inv = __powf(10000.0f, -(float)j * (1.0f / 512.0f));
  float ang = (float)s * inv;
  float sn, cs;
  __sincosf(ang, &sn, &cs);
  size_t i1 = ((size_t)(b * NH + h) * SEQ + s) * DK + dk;
  size_t i2 = ((size_t)(b * NH + h + 8) * SEQ + s) * DK + dk;
  float x1 = (float)arr[i1], x2 = (float)arr[i2];
  arr[i1] = (_Float16)(x1 * cs - x2 * sn);
  arr[i2] = (_Float16)(x1 * sn + x2 * cs);
}

// ---------------------------------------------------------------------------
// Kernel 3: banded attention, one workgroup per (block n, head h, batch b).
// Flash-style online softmax over the 3*256 key band in 64-key LDS chunks.
// K chunks stream in via async global->LDS copies (overlapped with the V
// register-transpose); OOB rows are clamped (finite data) and the mask
// predicate zeroes their probabilities exactly, so 0*finite = 0 in AV WMMA.
// ---------------------------------------------------------------------------
__global__ void attn(const _Float16* __restrict__ qh, const _Float16* __restrict__ kh,
                     const _Float16* __restrict__ vh, _Float16* __restrict__ ao) {
  __shared__ _Float16 Klds[64][80];           // K chunk, row-major [key][dk]
  __shared__ _Float16 Vt[64][80];             // V chunk, transposed [dk][key]
  __shared__ _Float16 Pb[8][2][16][32];       // per-wave P staging (C->A relayout)

  const int n = blockIdx.x, h = blockIdx.y, b = blockIdx.z;
  const int t = threadIdx.x, w = t >> 5, lane = t & 31;
  const int ncol = lane & 15, mhalf = (lane >> 4) << 3;

  const size_t headoff = (size_t)(b * NH + h) * SEQ * DK;
  const _Float16* qbase = qh + headoff + (size_t)n * BLK * DK;
  const _Float16* kbase = kh + headoff;
  const _Float16* vbase = vh + headoff;

  const int qlo = w * 32;
  v16h aQ[2][2];
#pragma unroll
  for (int mi = 0; mi < 2; ++mi)
#pragma unroll
    for (int kc = 0; kc < 2; ++kc)
      aQ[mi][kc] = frag_load(qbase + (qlo + mi * 16) * DK + kc * 32, DK);

  v8f acc[2][4];
  float mrow[2][8], lrow[2][8];
#pragma unroll
  for (int mi = 0; mi < 2; ++mi) {
#pragma unroll
    for (int dj = 0; dj < 4; ++dj) acc[mi][dj] = v8f_zero();
#pragma unroll
    for (int r = 0; r < 8; ++r) { mrow[mi][r] = NEG_INF; lrow[mi][r] = 0.f; }
  }

  const int band0 = (n - 1) * BLK;            // seq position of band index 0

  for (int ck = 0; ck < 12; ++ck) {           // 12 * 64 = 768 keys
#pragma unroll
    for (int it = 0; it < 2; ++it) {
      int u = t + it * 256;                   // 0..511 (16B units)
      int row = u >> 3, c = (u & 7) * 8;
      int seq = band0 + ck * 64 + row;
      seq = seq < 0 ? 0 : (seq > SEQ - 1 ? SEQ - 1 : seq);   // clamp; masked later
      // K: async copy straight into LDS (no VGPR round trip)
      ASYNC_COPY_B128(kbase + (size_t)seq * DK + c, &Klds[row][c]);
      // V: synchronous load + register transpose into Vt (overlaps async K)
      uint4 dV = *(const uint4*)(vbase + (size_t)seq * DK + c);
      const _Float16* pv = (const _Float16*)&dV;
#pragma unroll
      for (int j = 0; j < 8; ++j) Vt[c + j][row] = pv[j];
      // prefetch next chunk's lines while this one is consumed
      if (ck < 11) {
        int sq2 = band0 + (ck + 1) * 64 + row;
        sq2 = sq2 < 0 ? 0 : (sq2 > SEQ - 1 ? SEQ - 1 : sq2);
        __builtin_prefetch(kbase + (size_t)sq2 * DK + c, 0, 0);
        __builtin_prefetch(vbase + (size_t)sq2 * DK + c, 0, 0);
      }
    }
    ASYNC_WAIT();
    __syncthreads();

#pragma unroll
    for (int sub = 0; sub < 2; ++sub) {       // two 32-key WMMA K-steps
      const int k0 = ck * 64 + sub * 32;
      const int ki0 = k0 + ncol, ki1 = k0 + 16 + ncol;
      const int kp0 = band0 + ki0, kp1 = band0 + ki1;

#pragma unroll
      for (int mi = 0; mi < 2; ++mi) {
        v8f sc[2];
#pragma unroll
        for (int ni = 0; ni < 2; ++ni) {
          v8f c0 = v8f_zero();
          c0 = wmma_f16(aQ[mi][0], frag_load(&Klds[sub * 32 + ni * 16][0], 80), c0);
          c0 = wmma_f16(aQ[mi][1], frag_load(&Klds[sub * 32 + ni * 16][32], 80), c0);
          sc[ni] = c0;
        }
#pragma unroll
        for (int r = 0; r < 8; ++r) {
          int qi = qlo + mi * 16 + mhalf + r;
          int rel0 = ki0 - BLK - qi, rel1 = ki1 - BLK - qi;
          bool v0 = (rel0 >= -256) && (rel0 <= 256) && (kp0 >= 0) && (kp0 < SEQ);
          bool v1 = (rel1 >= -256) && (rel1 <= 256) && (kp1 >= 0) && (kp1 < SEQ);
          float a0 = sc[0][r] * 8.0f, a1 = sc[1][r] * 8.0f;  // * sqrt(dk)
          float s0 = v0 ? a0 : NEG_INF, s1 = v1 ? a1 : NEG_INF;
          float mx = fmaxf(s0, s1);
#pragma unroll
          for (int off = 1; off < 16; off <<= 1) mx = fmaxf(mx, __shfl_xor(mx, off, 16));
          float mold = mrow[mi][r];
          float mnew = fmaxf(mold, mx);
          float alpha = __expf(mold - mnew);
          float p0 = v0 ? __expf(a0 - mnew) : 0.f;
          float p1 = v1 ? __expf(a1 - mnew) : 0.f;
          float rs = p0 + p1;
#pragma unroll
          for (int off = 1; off < 16; off <<= 1) rs += __shfl_xor(rs, off, 16);
          mrow[mi][r] = mnew;
          lrow[mi][r] = lrow[mi][r] * alpha + rs;
#pragma unroll
          for (int dj = 0; dj < 4; ++dj) acc[mi][dj][r] *= alpha;
          Pb[w][mi][mhalf + r][ncol]      = (_Float16)p0;
          Pb[w][mi][mhalf + r][16 + ncol] = (_Float16)p1;
        }
      }
#pragma unroll
      for (int mi = 0; mi < 2; ++mi) {
        v16h aP = frag_load(&Pb[w][mi][0][0], 32);
#pragma unroll
        for (int dj = 0; dj < 4; ++dj)
          acc[mi][dj] = wmma_f16(aP, frag_load(&Vt[dj * 16][sub * 32], 80), acc[mi][dj]);
      }
    }
    __syncthreads();
  }

#pragma unroll
  for (int mi = 0; mi < 2; ++mi)
#pragma unroll
    for (int r = 0; r < 8; ++r) {
      float rinv = 1.0f / lrow[mi][r];        // every row has >=1 valid key
      int srow = n * BLK + qlo + mi * 16 + mhalf + r;
      size_t obase = ((size_t)b * SEQ + srow) * DIM + h * DK;
#pragma unroll
      for (int dj = 0; dj < 4; ++dj)
        ao[obase + dj * 16 + ncol] = (_Float16)(acc[mi][dj][r] * rinv);
    }
}

// ---------------------------------------------------------------------------
// Kernel 4: output projection.  out = AO * Wo^T + bias, fp32 result.
// A tile (already f16) streams in via async global->LDS; B tile converts
// fp32->f16 synchronously while the async copies are in flight.
// ---------------------------------------------------------------------------
__global__ void out_gemm(const _Float16* __restrict__ A, const float* __restrict__ W,
                         const float* __restrict__ bias, float* __restrict__ Out) {
  __shared__ _Float16 As[128][48];   // 96B pitch keeps b128 LDS dests 16B aligned
  __shared__ _Float16 Bs[64][40];
  const int t = threadIdx.x, w = t >> 5, lane = t & 31;
  const int ncol = lane & 15, mhalf = (lane >> 4) << 3;
  const int m0 = blockIdx.x * 128, n0 = blockIdx.y * 64;
  const int wm = (w >> 1) * 32, wn = (w & 1) * 32;

  v8f acc[2][2];
#pragma unroll
  for (int i = 0; i < 2; ++i)
#pragma unroll
    for (int j = 0; j < 2; ++j) acc[i][j] = v8f_zero();

  for (int k0 = 0; k0 < DIM; k0 += 32) {
    // A tile: 128x32 f16, async (512 x 16B, 2 per thread)
#pragma unroll
    for (int i = 0; i < 2; ++i) {
      int u = t + i * 256;
      int row = u >> 2, c8 = (u & 3) * 8;
      ASYNC_COPY_B128(A + (size_t)(m0 + row) * DIM + k0 + c8, &As[row][c8]);
    }
    // B tile: fp32 -> f16 (overlaps async A)
#pragma unroll
    for (int i = 0; i < 2; ++i) {
      int u = t + i * 256;
      int row = u >> 3, c4 = (u & 7) * 4;
      float4 d = *(const float4*)(W + (size_t)(n0 + row) * DIM + k0 + c4);
      Bs[row][c4 + 0] = (_Float16)d.x;  Bs[row][c4 + 1] = (_Float16)d.y;
      Bs[row][c4 + 2] = (_Float16)d.z;  Bs[row][c4 + 3] = (_Float16)d.w;
    }
    ASYNC_WAIT();
    __syncthreads();

    v16h a0 = frag_load(&As[wm][0], 48);
    v16h a1 = frag_load(&As[wm + 16][0], 48);
    v16h b0 = frag_load(&Bs[wn][0], 40);
    v16h b1 = frag_load(&Bs[wn + 16][0], 40);
    acc[0][0] = wmma_f16(a0, b0, acc[0][0]);
    acc[0][1] = wmma_f16(a0, b1, acc[0][1]);
    acc[1][0] = wmma_f16(a1, b0, acc[1][0]);
    acc[1][1] = wmma_f16(a1, b1, acc[1][1]);
    __syncthreads();
  }

#pragma unroll
  for (int i = 0; i < 2; ++i)
#pragma unroll
    for (int j = 0; j < 2; ++j)
#pragma unroll
      for (int r = 0; r < 8; ++r) {
        int m = m0 + wm + i * 16 + mhalf + r;
        int e = n0 + wn + j * 16 + ncol;
        Out[(size_t)m * DIM + e] = acc[i][j][r] + bias[e];
      }
}

// ---------------------------------------------------------------------------
extern "C" void kernel_launch(void* const* d_in, const int* in_sizes, int n_in,
                              void* d_out, int out_size, void* d_ws, size_t ws_size,
                              hipStream_t stream) {
  const float* q  = (const float*)d_in[0];
  const float* k  = (const float*)d_in[1];
  const float* v  = (const float*)d_in[2];
  const float* wq = (const float*)d_in[3];
  const float* wk = (const float*)d_in[4];
  const float* wv = (const float*)d_in[5];
  const float* wo = (const float*)d_in[6];
  const float* wb = (const float*)d_in[7];

  const size_t HEADELTS = (size_t)2 * NH * SEQ * DK;   // 8,388,608 halfs = 16 MB
  _Float16* qh = (_Float16*)d_ws;
  _Float16* kh = qh + HEADELTS;
  _Float16* vh = kh + HEADELTS;
  _Float16* ao = vh + HEADELTS;                        // [b][s][d] f16

  dim3 gproj(64, 16);                                  // 8192/128 x 1024/64
  proj_gemm<<<gproj, 256, 0, stream>>>(q, wq, qh);
  proj_gemm<<<gproj, 256, 0, stream>>>(k, wk, kh);
  proj_gemm<<<gproj, 256, 0, stream>>>(v, wv, vh);

  rope_inplace<<<(2 * 8 * SEQ * DK) / 256, 256, 0, stream>>>(qh);
  rope_inplace<<<(2 * 8 * SEQ * DK) / 256, 256, 0, stream>>>(kh);

  attn<<<dim3(SEQ / BLK, NH, 2), 256, 0, stream>>>(qh, kh, vh, ao);

  out_gemm<<<gproj, 256, 0, stream>>>(ao, wo, wb, (float*)d_out);
}
